// SpatioTemporalGNNEmbedder_13451837571914
// MI455X (gfx1250) — matrix operand order repaired and checked by reading, not compiled, not run on hardware
//
#include <hip/hip_runtime.h>
#include <hip/hip_bf16.h>

// ---------------------------------------------------------------------------
// SpatioTemporalGNNEmbedder for gfx1250 (MI455X, wave32, WMMA)
//   LSTM (fused, WMMA gate GEMM, weights register-resident) ->
//   per-timestep GATv2 x2 (WMMA GEMMs + atomic scatter softmax/aggregate) ->
//   BN/ReLU -> WMMA projection.
// All matmuls use v_wmma_f32_16x16x32_bf16 with hand-placed fragments per
// CDNA5 ISA 7.12.2. Scatter uses native f32 global atomics (L2-resident data).
// LSTM uses 32 waves x 1 column-tile each so the loop-invariant weight
// fragments (5 x 8 = 40 VGPRs) stay register-resident across the t-loop
// (no spill, no forced reload).
// ---------------------------------------------------------------------------

#define NNODES 20000
#define TSTEPS 8
#define HCC    512      // GAT concat channels (4 heads x 128)
#define GHH    4
#define GCC    128
#define EMBD   64
#define K1     160      // padded GNN layer-1 K (16 static + 128 lstm -> 160)
#define KL     160      // padded LSTM input K (8 x + 128 h -> 160)

typedef __attribute__((ext_vector_type(16))) __bf16 v16bf;
typedef __attribute__((ext_vector_type(8)))  float  v8f;

union BF16Frag { unsigned u[8]; v16bf v; };

// ---- WMMA fragment loaders (CDNA5 ISA 7.12.2 16-bit layouts) ---------------
// A (16x32, MxK): lane l holds row m = l&15; half = l>>4;
//   VGPR v holds K = (v>>2)*16 + half*8 + (v&3)*2 (+1)
__device__ inline v16bf load_a_frag(const __bf16* A, int lda, int m0, int k0, int lane) {
  const __bf16* row = A + (size_t)(m0 + (lane & 15)) * lda;
  int half = lane >> 4;
  BF16Frag f;
#pragma unroll
  for (int v = 0; v < 8; ++v) {
    int k = k0 + ((v >> 2) << 4) + (half << 3) + ((v & 3) << 1);
    f.u[v] = *(const unsigned*)(row + k);
  }
  return f.v;
}
// B (32x16, KxN), sourced from weight W[ncols][K] row-major (B = W^T):
//   lane l holds column n = l&15; half = l>>4; VGPR v holds K = (v>>2)*16 + half*8 + (v&3)*2 (+1)
__device__ inline v16bf load_b_frag(const __bf16* W, int ldw, int n0, int k0, int lane) {
  const __bf16* row = W + (size_t)(n0 + (lane & 15)) * ldw;
  int half = lane >> 4;
  BF16Frag f;
#pragma unroll
  for (int v = 0; v < 8; ++v) {
    int k = k0 + ((v >> 2) << 4) + (half << 3) + ((v & 3) << 1);
    f.u[v] = *(const unsigned*)(row + k);
  }
  return f.v;
}

__device__ inline float sigmoidf(float x) { return 1.f / (1.f + __expf(-x)); }

// monotonic float <-> uint mapping so atomicMax(uint) implements float max
__device__ inline unsigned fmapu(float f) {
  unsigned u = __float_as_uint(f);
  return (u & 0x80000000u) ? ~u : (u | 0x80000000u);
}
__device__ inline float funmap(unsigned u) {
  return __uint_as_float((u & 0x80000000u) ? (u ^ 0x80000000u) : ~u);
}
__device__ inline void atomAddF(float* p, float v) {
  __hip_atomic_fetch_add(p, v, __ATOMIC_RELAXED, __HIP_MEMORY_SCOPE_AGENT);
}

// ---- weight prep: f32 -> bf16 with K padding -------------------------------
__global__ void cvt_pad(const float* __restrict__ src, __bf16* __restrict__ dst,
                        int rows, int ks, int kd) {
  int idx = blockIdx.x * blockDim.x + threadIdx.x;
  if (idx >= rows * kd) return;
  int r = idx / kd, k = idx - r * kd;
  dst[idx] = (k < ks) ? (__bf16)src[(size_t)r * ks + k] : (__bf16)0.f;
}
// Wcat[512][160] = [W_ih (K=8) | W_hh (K=128) | 0 pad]
__global__ void cvt_cat(const float* __restrict__ Wih, const float* __restrict__ Whh,
                        __bf16* __restrict__ dst) {
  int idx = blockIdx.x * blockDim.x + threadIdx.x;
  if (idx >= 512 * KL) return;
  int r = idx / KL, k = idx - r * KL;
  float v = 0.f;
  if (k < 8)        v = Wih[(size_t)r * 8 + k];
  else if (k < 136) v = Whh[(size_t)r * 128 + (k - 8)];
  dst[idx] = (__bf16)v;
}

// ---- fused LSTM: 16 nodes/block, 32 waves x 1 tile, full T loop ------------
__global__ void __launch_bounds__(1024)
lstm_kernel(const float* __restrict__ xs, const float* __restrict__ dyn,
            const __bf16* __restrict__ Wcat,
            const float* __restrict__ b_ih, const float* __restrict__ b_hh,
            __bf16* __restrict__ A1) {
  __shared__ __bf16 uA[16][KL];     // [x_t(8) | h(128) | pad]
  __shared__ float  gL[16][512];    // raw gates
  __shared__ float  cL[16][128];    // cell state
  int tid = threadIdx.x, lane = tid & 31, wave = tid >> 5;   // 32 waves
  int n0 = blockIdx.x * 16;

  for (int i = tid; i < 16 * KL; i += 1024) (&uA[0][0])[i] = (__bf16)0.f;
  for (int i = tid; i < 16 * 128; i += 1024) (&cL[0][0])[i] = 0.f;

  // this wave's 16 gate columns: preload all 5 K-step weight fragments into
  // registers once (40 VGPRs), reused for all 8 timesteps
  v16bf bfr[5];
#pragma unroll
  for (int kk = 0; kk < 5; ++kk)
    bfr[kk] = load_b_frag(Wcat, KL, wave * 16, kk * 32, lane);

  for (int t = 0; t < TSTEPS; ++t) {
    // stage x_t into uA[:, 0:8]
    if (tid < 16 * 8) {
      int m = tid >> 3, k = tid & 7;
      uA[m][k] = (__bf16)dyn[(size_t)(n0 + m) * (TSTEPS * 8) + t * 8 + k];
    }
    __syncthreads();

    // gates[:, wave*16 : wave*16+16] = u @ Wcat^T, K = 160
    v8f acc = {};
#pragma unroll
    for (int kk = 0; kk < 5; ++kk) {
      v16bf a = load_a_frag(&uA[0][0], KL, 0, kk * 32, lane);
      acc = __builtin_amdgcn_wmma_f32_16x16x32_bf16(false, a, false, bfr[kk],
                                                    (short)0, acc, false, false);
    }
    int mrow = (lane >> 4) * 8, nn = lane & 15;
#pragma unroll
    for (int r = 0; r < 8; ++r)
      gL[mrow + r][wave * 16 + nn] = acc[r];
    __syncthreads();

    // elementwise gate math, state update, emit h (bf16) to uA and A1
    for (int i = tid; i < 16 * 128; i += 1024) {
      int m = i >> 7, k = i & 127;
      float gi = sigmoidf(gL[m][k]       + b_ih[k]       + b_hh[k]);
      float gf = sigmoidf(gL[m][128 + k] + b_ih[128 + k] + b_hh[128 + k]);
      float gg = tanhf  (gL[m][256 + k]  + b_ih[256 + k] + b_hh[256 + k]);
      float go = sigmoidf(gL[m][384 + k] + b_ih[384 + k] + b_hh[384 + k]);
      float c = gf * cL[m][k] + gi * gg;
      cL[m][k] = c;
      float h = go * tanhf(c);
      uA[m][8 + k] = (__bf16)h;
      A1[((size_t)t * NNODES + n0 + m) * K1 + 16 + k] = (__bf16)h;
    }
    // A1 static part + zero pad
    if (tid < 16 * 32) {
      int m = tid >> 5, k = tid & 31;
      __bf16 v = (k < 16) ? (__bf16)xs[(size_t)(n0 + m) * 16 + k] : (__bf16)0.f;
      A1[((size_t)t * NNODES + n0 + m) * K1 + (k < 16 ? k : 128 + k)] = v;
    }
    __syncthreads();
  }
}

// ---- generic WMMA GEMM: C[M x (WAVES*TILES*16)] = A[M x K] @ W[N x K]^T ----
template <int WAVES, int TILES>
__global__ void gemm_bf16(const __bf16* __restrict__ A, int lda,
                          const __bf16* __restrict__ W, int ldw,
                          float* __restrict__ C, int ldc,
                          const float* __restrict__ bias, int K) {
  int lane = threadIdx.x & 31, wave = threadIdx.x >> 5;
  int m0 = blockIdx.x * 16;
  v8f acc[TILES] = {};
  for (int k0 = 0; k0 < K; k0 += 32) {
    v16bf a = load_a_frag(A, lda, m0, k0, lane);
#pragma unroll
    for (int j = 0; j < TILES; ++j) {
      v16bf b = load_b_frag(W, ldw, (wave * TILES + j) * 16, k0, lane);
      acc[j] = __builtin_amdgcn_wmma_f32_16x16x32_bf16(false, a, false, b,
                                                       (short)0, acc[j], false, false);
    }
  }
  int mrow = (lane >> 4) * 8, n = lane & 15;
#pragma unroll
  for (int j = 0; j < TILES; ++j) {
    int n0 = (wave * TILES + j) * 16;
    float bv = bias ? bias[n0 + n] : 0.f;
#pragma unroll
    for (int r = 0; r < 8; ++r)
      C[(size_t)(m0 + mrow + r) * ldc + n0 + n] = acc[j][r] + bv;
  }
}

// ---- per-layer init: out = bias, smax = -inf(map), den = 0, stats = 0 ------
__global__ void init_layer(float* __restrict__ out, const float* __restrict__ bias,
                           unsigned* __restrict__ smax, float* __restrict__ den,
                           float* __restrict__ colsum, float* __restrict__ colsq) {
  int idx = blockIdx.x * blockDim.x + threadIdx.x;
  if (idx < NNODES * HCC) out[idx] = bias[idx & (HCC - 1)];
  if (idx < NNODES * GHH) { smax[idx] = fmapu(-__builtin_inff()); den[idx] = 0.f; }
  if (idx < HCC) { colsum[idx] = 0.f; colsq[idx] = 0.f; }
}

// ---- edge score: s[e][h] = att[h].dot(leaky(xl[src]+xr[dst])); segment max -
__global__ void edge_score(const float* __restrict__ xl, const float* __restrict__ xr,
                           const float* __restrict__ att, const int* __restrict__ ei,
                           float* __restrict__ s, unsigned* __restrict__ smax, int E) {
  int lane = threadIdx.x & 31, wave = threadIdx.x >> 5;
  int e = blockIdx.x * 8 + wave;
  int Etot = E + NNODES;
  if (e >= Etot) return;
  int src = (e < E) ? ei[e] : (e - E);
  int dst = (e < E) ? ei[E + e] : (e - E);
  int c0 = lane * 16, head = lane >> 3;
  const float4* pl = (const float4*)(xl + (size_t)src * HCC + c0);
  const float4* pr = (const float4*)(xr + (size_t)dst * HCC + c0);
  const float4* pa = (const float4*)(att + head * GCC + (c0 & (GCC - 1)));
  float partial = 0.f;
#pragma unroll
  for (int q = 0; q < 4; ++q) {
    float4 a = pl[q], b = pr[q], w = pa[q];
    float v;
    v = a.x + b.x; v = v > 0.f ? v : 0.2f * v; partial += w.x * v;
    v = a.y + b.y; v = v > 0.f ? v : 0.2f * v; partial += w.y * v;
    v = a.z + b.z; v = v > 0.f ? v : 0.2f * v; partial += w.z * v;
    v = a.w + b.w; v = v > 0.f ? v : 0.2f * v; partial += w.w * v;
  }
  partial += __shfl_xor(partial, 1);
  partial += __shfl_xor(partial, 2);
  partial += __shfl_xor(partial, 4);
  if ((lane & 7) == 0) {
    s[(size_t)e * GHH + head] = partial;
    atomicMax(&smax[(size_t)dst * GHH + head], fmapu(partial));
  }
}

// ---- edge exp: ex = exp(s - smax[dst]); den[dst] += ex ---------------------
__global__ void edge_exp(float* __restrict__ s, const unsigned* __restrict__ smax,
                         float* __restrict__ den, const int* __restrict__ ei, int E) {
  int idx = blockIdx.x * blockDim.x + threadIdx.x;
  int Etot = E + NNODES;
  if (idx >= Etot * GHH) return;
  int e = idx >> 2, h = idx & 3;
  int dst = (e < E) ? ei[E + e] : (e - E);
  float ex = __expf(s[idx] - funmap(smax[(size_t)dst * GHH + h]));
  s[idx] = ex;
  atomAddF(&den[(size_t)dst * GHH + h], ex);
}

// ---- edge aggregate: out[dst] += (ex/den[dst]) * xl[src] -------------------
__global__ void edge_agg(const float* __restrict__ xl, const float* __restrict__ s,
                         const float* __restrict__ den, const int* __restrict__ ei,
                         float* __restrict__ out, int E) {
  int lane = threadIdx.x & 31, wave = threadIdx.x >> 5;
  int e = blockIdx.x * 8 + wave;
  int Etot = E + NNODES;
  if (e >= Etot) return;
  int src = (e < E) ? ei[e] : (e - E);
  int dst = (e < E) ? ei[E + e] : (e - E);
  int c0 = lane * 16, head = lane >> 3;
  float alpha = s[(size_t)e * GHH + head] / den[(size_t)dst * GHH + head];
  const float4* pl = (const float4*)(xl + (size_t)src * HCC + c0);
  float* po = out + (size_t)dst * HCC + c0;
#pragma unroll
  for (int q = 0; q < 4; ++q) {
    float4 a = pl[q];
    atomAddF(po + q * 4 + 0, alpha * a.x);
    atomAddF(po + q * 4 + 1, alpha * a.y);
    atomAddF(po + q * 4 + 2, alpha * a.z);
    atomAddF(po + q * 4 + 3, alpha * a.w);
  }
}

// ---- BN stats: per-column sum / sumsq over 64-row strips -------------------
__global__ void bn_stats(const float* __restrict__ out, float* __restrict__ colsum,
                         float* __restrict__ colsq) {
  int c = threadIdx.x;                 // columns c and c+256
  int r0 = blockIdx.x * 64;
  float s0 = 0.f, q0 = 0.f, s1 = 0.f, q1 = 0.f;
  for (int r = 0; r < 64; ++r) {
    int row = r0 + r;
    if (row >= NNODES) break;
    float a = out[(size_t)row * HCC + c];       s0 += a; q0 += a * a;
    float b = out[(size_t)row * HCC + c + 256]; s1 += b; q1 += b * b;
  }
  atomAddF(&colsum[c], s0);       atomAddF(&colsq[c], q0);
  atomAddF(&colsum[c + 256], s1); atomAddF(&colsq[c + 256], q1);
}

// ---- BN apply + ReLU -> bf16 activation for next GEMM ----------------------
__global__ void bn_apply(const float* __restrict__ out, const float* __restrict__ colsum,
                         const float* __restrict__ colsq, const float* __restrict__ gamma,
                         const float* __restrict__ beta, __bf16* __restrict__ dst) {
  int idx = blockIdx.x * blockDim.x + threadIdx.x;
  if (idx >= NNODES * HCC) return;
  int c = idx & (HCC - 1);
  const float invN = 1.f / NNODES;
  float m = colsum[c] * invN;
  float v = colsq[c] * invN - m * m;
  float y = (out[idx] - m) * rsqrtf(v + 1e-5f) * gamma[c] + beta[c];
  dst[idx] = (__bf16)(y > 0.f ? y : 0.f);
}

// ---------------------------------------------------------------------------
extern "C" void kernel_launch(void* const* d_in, const int* in_sizes, int n_in,
                              void* d_out, int out_size, void* d_ws, size_t ws_size,
                              hipStream_t stream) {
  const float* xs    = (const float*)d_in[0];
  const float* dyn   = (const float*)d_in[1];
  const int*   ei    = (const int*)  d_in[2];
  const float* W_ih  = (const float*)d_in[3];
  const float* W_hh  = (const float*)d_in[4];
  const float* b_ih  = (const float*)d_in[5];
  const float* b_hh  = (const float*)d_in[6];
  const float* Wl1   = (const float*)d_in[7];
  const float* Wr1   = (const float*)d_in[8];
  const float* att1  = (const float*)d_in[9];
  const float* bg1   = (const float*)d_in[10];
  const float* Wl2   = (const float*)d_in[11];
  const float* Wr2   = (const float*)d_in[12];
  const float* att2  = (const float*)d_in[13];
  const float* bg2   = (const float*)d_in[14];
  const float* gam1  = (const float*)d_in[15];
  const float* bet1  = (const float*)d_in[16];
  const float* gam2  = (const float*)d_in[17];
  const float* bet2  = (const float*)d_in[18];
  const float* Wp    = (const float*)d_in[19];
  const float* bp    = (const float*)d_in[20];
  float* dout = (float*)d_out;

  const int E = in_sizes[2] / 2;
  const int Etot = E + NNODES;

  // workspace carve-out (~225 MB)
  char* p = (char*)d_ws;
  auto take = [&](size_t bytes) { char* r = p; p += (bytes + 255) & ~(size_t)255; return r; };
  __bf16* Wcat = (__bf16*)take((size_t)512 * KL * 2);
  __bf16* Wl1p = (__bf16*)take((size_t)512 * K1 * 2);
  __bf16* Wr1p = (__bf16*)take((size_t)512 * K1 * 2);
  __bf16* Wl2p = (__bf16*)take((size_t)512 * 512 * 2);
  __bf16* Wr2p = (__bf16*)take((size_t)512 * 512 * 2);
  __bf16* Wpp  = (__bf16*)take((size_t)64 * 512 * 2);
  __bf16* A1   = (__bf16*)take((size_t)TSTEPS * NNODES * K1 * 2);
  __bf16* A2   = (__bf16*)take((size_t)NNODES * HCC * 2);
  __bf16* A3   = (__bf16*)take((size_t)NNODES * HCC * 2);
  float*  xlb  = (float*) take((size_t)NNODES * HCC * 4);
  float*  xrb  = (float*) take((size_t)NNODES * HCC * 4);
  float*  outb = (float*) take((size_t)NNODES * HCC * 4);
  float*  sbuf = (float*) take((size_t)Etot * GHH * 4);
  unsigned* smax = (unsigned*)take((size_t)NNODES * GHH * 4);
  float*  den  = (float*) take((size_t)NNODES * GHH * 4);
  float*  csum = (float*) take((size_t)HCC * 4);
  float*  csq  = (float*) take((size_t)HCC * 4);

  const int MB = NNODES / 16;                 // 1250 row blocks
  const int eBlocks = (Etot + 7) / 8;
  const int nodeElems = NNODES * HCC;

  // weight prep (bf16, K padded to multiples of 32)
  cvt_cat<<<(512 * KL + 255) / 256, 256, 0, stream>>>(W_ih, W_hh, Wcat);
  cvt_pad<<<(512 * K1 + 255) / 256, 256, 0, stream>>>(Wl1, Wl1p, 512, 144, K1);
  cvt_pad<<<(512 * K1 + 255) / 256, 256, 0, stream>>>(Wr1, Wr1p, 512, 144, K1);
  cvt_pad<<<(512 * 512 + 255) / 256, 256, 0, stream>>>(Wl2, Wl2p, 512, 512, 512);
  cvt_pad<<<(512 * 512 + 255) / 256, 256, 0, stream>>>(Wr2, Wr2p, 512, 512, 512);
  cvt_pad<<<(64 * 512 + 255) / 256, 256, 0, stream>>>(Wp, Wpp, 64, 512, 512);

  // fused LSTM over all T; emits padded bf16 layer-1 inputs A1[t]
  lstm_kernel<<<MB, 1024, 0, stream>>>(xs, dyn, Wcat, b_ih, b_hh, A1);

  for (int t = 0; t < TSTEPS; ++t) {
    const __bf16* At = A1 + (size_t)t * NNODES * K1;

    // ---- GAT layer 1 ----
    gemm_bf16<8, 4><<<MB, 256, 0, stream>>>(At, K1, Wl1p, K1, xlb, HCC, nullptr, K1);
    gemm_bf16<8, 4><<<MB, 256, 0, stream>>>(At, K1, Wr1p, K1, xrb, HCC, nullptr, K1);
    init_layer<<<(nodeElems + 255) / 256, 256, 0, stream>>>(outb, bg1, smax, den, csum, csq);
    edge_score<<<eBlocks, 256, 0, stream>>>(xlb, xrb, att1, ei, sbuf, smax, E);
    edge_exp<<<(Etot * GHH + 255) / 256, 256, 0, stream>>>(sbuf, smax, den, ei, E);
    edge_agg<<<eBlocks, 256, 0, stream>>>(xlb, sbuf, den, ei, outb, E);
    bn_stats<<<(NNODES + 63) / 64, 256, 0, stream>>>(outb, csum, csq);
    bn_apply<<<(nodeElems + 255) / 256, 256, 0, stream>>>(outb, csum, csq, gam1, bet1, A2);

    // ---- GAT layer 2 ----
    gemm_bf16<8, 4><<<MB, 256, 0, stream>>>(A2, 512, Wl2p, 512, xlb, HCC, nullptr, 512);
    gemm_bf16<8, 4><<<MB, 256, 0, stream>>>(A2, 512, Wr2p, 512, xrb, HCC, nullptr, 512);
    init_layer<<<(nodeElems + 255) / 256, 256, 0, stream>>>(outb, bg2, smax, den, csum, csq);
    edge_score<<<eBlocks, 256, 0, stream>>>(xlb, xrb, att2, ei, sbuf, smax, E);
    edge_exp<<<(Etot * GHH + 255) / 256, 256, 0, stream>>>(sbuf, smax, den, ei, E);
    edge_agg<<<eBlocks, 256, 0, stream>>>(xlb, sbuf, den, ei, outb, E);
    bn_stats<<<(NNODES + 63) / 64, 256, 0, stream>>>(outb, csum, csq);
    bn_apply<<<(nodeElems + 255) / 256, 256, 0, stream>>>(outb, csum, csq, gam2, bet2, A3);

    // ---- projection into d_out[n][t][0:64] (row stride T*EMB = 512) ----
    gemm_bf16<4, 1><<<MB, 128, 0, stream>>>(A3, 512, Wpp, 512,
                                            dout + (size_t)t * EMBD, TSTEPS * EMBD, bp, 512);
  }
}
